// MessagePassingBlock_51608327029104
// MI455X (gfx1250) — matrix operand order repaired and checked by reading, not compiled, not run on hardware
//
#include <hip/hip_runtime.h>
#include <hip/hip_bf16.h>

typedef __attribute__((ext_vector_type(16))) _Float16 v16h;
typedef __attribute__((ext_vector_type(8)))  _Float16 v8h;
typedef __attribute__((ext_vector_type(8)))  float    v8f;

#define C_DIM 128
#define H_DIM 90
#define W_DIM 160
#define B_DIM 8
#define CH_STRIDE (H_DIM * W_DIM)          // 14400
#define KFRAGS 36                           // K = 9*128 = 1152 -> 36 frags of 32
#define WFRAG_HALFS (8 * KFRAGS * 32 * 16)  // per-direction A-frag array (147456 halfs = 288 KB)

// ---------------------------------------------------------------------------
// Pre-swizzle conv weights [O=128, I=128, K=9] (f32) into WMMA A-fragment
// layout (f16), tap-major K ordering: Kglob = kk*128 + i.
// A 16x32 layout (contiguous K, matches B): lane<16: elems 0..15 -> K=0..15
//                                          lane>=16: elems 0..15 -> K=16..31
// ---------------------------------------------------------------------------
__global__ void prep_weights(const float* __restrict__ w0, const float* __restrict__ w1,
                             const float* __restrict__ w2, const float* __restrict__ w3,
                             _Float16* __restrict__ frags) {
  int slot = blockIdx.x * blockDim.x + threadIdx.x;
  const int total = 4 * 8 * KFRAGS * 32;
  if (slot >= total) return;
  int d     = slot / (8 * KFRAGS * 32);
  int rem   = slot % (8 * KFRAGS * 32);
  int otile = rem / (KFRAGS * 32);
  int kf    = (rem / 32) % KFRAGS;
  int lane  = rem & 31;
  const float* w = (d == 0) ? w0 : (d == 1) ? w1 : (d == 2) ? w2 : w3;
  int m = otile * 16 + (lane & 15);
  _Float16* dst = frags + (size_t)d * WFRAG_HALFS +
                  ((size_t)(otile * KFRAGS + kf) * 32 + lane) * 16;
#pragma unroll
  for (int e = 0; e < 16; ++e) {
    int K  = kf * 32 + e + ((lane & 16) ? 16 : 0);
    int i  = K & 127;   // input channel
    int kk = K >> 7;    // tap 0..8
    dst[e] = (_Float16)w[((size_t)m * C_DIM + i) * 9 + kk];
  }
}

// out = x (initialize accumulator; each direction kernel then adds its h-scan)
__global__ void init_out(const float4* __restrict__ x, float4* __restrict__ out, int n4) {
  int i = blockIdx.x * blockDim.x + threadIdx.x;
  if (i < n4) out[i] = x[i];
}

// ---------------------------------------------------------------------------
// One directional SCNN scan for one batch per workgroup.
//   MDIM  = conv-axis length (160 vertical / 90 horizontal), NT tiles of 16
//   L     = scan length (90 vertical / 160 horizontal)
//   addr(c, l, m) = base + c*CH_STRIDE + l*scan_stride + m*conv_stride
// h_prev kept in LDS as f16, channel-contiguous, 4-col zero halo each side.
// 8 waves, wave w owns output-channel tile w; NT f32 accumulators per wave.
// ---------------------------------------------------------------------------
template <int MDIM>
__global__ __launch_bounds__(256) void scan_dir(const float* __restrict__ x,
                                                float* __restrict__ out,
                                                const _Float16* __restrict__ wfrag,
                                                const float* __restrict__ bias,
                                                int L, int scan_stride, int conv_stride,
                                                int rev) {
  constexpr int NT   = (MDIM + 15) / 16;
  constexpr int COLS = NT * 16 + 8;  // +4 halo each side (zero-padded 'same' conv)
  __shared__ _Float16 hlds[COLS * C_DIM];

  const int tid  = threadIdx.x;
  const int wave = tid >> 5;
  const int lane = tid & 31;
  const size_t base = (size_t)blockIdx.x * C_DIM * CH_STRIDE;

  // zero LDS (halo columns + ragged tail for MDIM=90 stay zero forever)
  for (int i = tid; i < COLS * C_DIM; i += 256) hlds[i] = (_Float16)0.f;
  __syncthreads();

  // step 0: h0 = x[l0];  out += h0
  {
    int l = rev ? (L - 1) : 0;
    for (int idx = tid; idx < C_DIM * MDIM; idx += 256) {
      int c = idx / MDIM, m = idx % MDIM;
      size_t ga = base + (size_t)c * CH_STRIDE + (size_t)l * scan_stride +
                  (size_t)m * conv_stride;
      float v = x[ga];
      hlds[(m + 4) * C_DIM + c] = (_Float16)v;
      out[ga] += v;
    }
  }
  __syncthreads();

  const int otile  = wave;
  const int o_base = otile * 16 + ((lane & 16) ? 8 : 0);  // f32 C/D: upper lanes = M+8
  float bv[8];
#pragma unroll
  for (int r = 0; r < 8; ++r) bv[r] = bias[o_base + r];
  const _Float16* wbase = wfrag + (size_t)otile * KFRAGS * 512 + (size_t)lane * 16;

  for (int step = 1; step < L; ++step) {
    int l = rev ? (L - 1 - step) : step;
    v8f acc[NT];
#pragma unroll
    for (int nt = 0; nt < NT; ++nt) {
      v8f z = {0.f, 0.f, 0.f, 0.f, 0.f, 0.f, 0.f, 0.f};
      acc[nt] = z;
    }

    // GEMM: conv[o, n] = sum_K A(o,K) * B(K,n), K = kk*128 + i
    for (int kf = 0; kf < KFRAGS; ++kf) {
      v16h a = *(const v16h*)(wbase + (size_t)kf * 512);  // L2-resident A frag
      int kk = kf >> 2;                                   // tap for this K-frag
      int i0 = ((kf & 3) << 5) + ((lane & 16) ? 16 : 0);  // channel base (B layout)
#pragma unroll
      for (int nt = 0; nt < NT; ++nt) {
        int n = nt * 16 + (lane & 15);
        v16h b = *(const v16h*)&hlds[(n + kk) * C_DIM + i0];  // halo handles pad
        acc[nt] = __builtin_amdgcn_wmma_f32_16x16x32_f16(
            false, a, false, b, (short)0, acc[nt], false, false);
      }
    }

    __syncthreads();  // all waves done reading h_prev before overwrite

    // epilogue: h = x + relu(conv + bias); out += h; h -> LDS (f16)
#pragma unroll
    for (int nt = 0; nt < NT; ++nt) {
      int n = nt * 16 + (lane & 15);
      if (n < MDIM) {
        size_t ga0 = base + (size_t)o_base * CH_STRIDE + (size_t)l * scan_stride +
                     (size_t)n * conv_stride;
        v8h hv;
#pragma unroll
        for (int r = 0; r < 8; ++r) {
          float v = acc[nt][r] + bv[r];
          v = v > 0.f ? v : 0.f;
          size_t ga = ga0 + (size_t)r * CH_STRIDE;
          float h = x[ga] + v;
          out[ga] += h;
          hv[r] = (_Float16)h;
        }
        *(v8h*)&hlds[(n + 4) * C_DIM + o_base] = hv;  // 16B packed LDS store
      }
    }
    __syncthreads();  // h updated before next step's B reads
  }
}

extern "C" void kernel_launch(void* const* d_in, const int* in_sizes, int n_in,
                              void* d_out, int out_size, void* d_ws, size_t ws_size,
                              hipStream_t stream) {
  const float* x       = (const float*)d_in[0];
  const float* up_w    = (const float*)d_in[1];
  const float* up_b    = (const float*)d_in[2];
  const float* down_w  = (const float*)d_in[3];
  const float* down_b  = (const float*)d_in[4];
  const float* left_w  = (const float*)d_in[5];
  const float* left_b  = (const float*)d_in[6];
  const float* right_w = (const float*)d_in[7];
  const float* right_b = (const float*)d_in[8];
  float* out = (float*)d_out;
  _Float16* frags = (_Float16*)d_ws;  // 4 * 288 KB = 1.125 MB

  const int slots = 4 * 8 * KFRAGS * 32;
  prep_weights<<<(slots + 255) / 256, 256, 0, stream>>>(up_w, down_w, left_w, right_w, frags);

  const int n4 = B_DIM * C_DIM * CH_STRIDE / 4;
  init_out<<<(n4 + 255) / 256, 256, 0, stream>>>((const float4*)x, (float4*)out, n4);

  // vertical scans: conv over W (MDIM=160), recurrence over H (L=90)
  scan_dir<160><<<B_DIM, 256, 0, stream>>>(x, out, frags + 0 * (size_t)WFRAG_HALFS, up_b,
                                           H_DIM, W_DIM, 1, 0);  // top -> bottom
  scan_dir<160><<<B_DIM, 256, 0, stream>>>(x, out, frags + 1 * (size_t)WFRAG_HALFS, down_b,
                                           H_DIM, W_DIM, 1, 1);  // bottom -> top
  // horizontal scans: conv over H (MDIM=90), recurrence over W (L=160)
  scan_dir<90><<<B_DIM, 256, 0, stream>>>(x, out, frags + 2 * (size_t)WFRAG_HALFS, left_b,
                                          W_DIM, 1, W_DIM, 0);  // left -> right
  scan_dir<90><<<B_DIM, 256, 0, stream>>>(x, out, frags + 3 * (size_t)WFRAG_HALFS, right_b,
                                          W_DIM, 1, W_DIM, 1);  // right -> left
}
